// Mambas_22797686408077
// MI455X (gfx1250) — matrix operand order repaired and checked by reading, not compiled, not run on hardware
//
#include <hip/hip_runtime.h>
#include <hip/hip_bf16.h>

// MI455X / gfx1250, wave32. WMMA bf16 (16x16x32) GEMMs + register-resident
// selective scan + split-K WMMA for the 268MB W_fc reduction.

#define B_  16
#define L_  1024
#define DM_ 512
#define DI_ 1024
#define ST_ 16
#define DC_ 4
#define RK_ 32
#define OD_ 128
#define ROWS_ (B_ * L_)     // 16384

typedef __attribute__((ext_vector_type(16))) __bf16 v16bf;
typedef __attribute__((ext_vector_type(8)))  float  v8f;

// ---------------------------------------------------------------------------
// Tiled bf16 WMMA GEMM.  Each wave computes a 32x32 output patch (2x2 wmma
// tiles); block = WM_TILES x WN_TILES waves -> (32*WM_TILES) x (32*WN_TILES)
// block tile.  K advanced in chunks of 32 through LDS (fp32 -> bf16 on stage).
// EPI: 0 = plain store, 1 = softplus(acc + bias[col])  (for the dt branch).
// ---------------------------------------------------------------------------
template<int WM_TILES, int WN_TILES, int EPI>
__global__ void gemm_bf16_kernel(const float* __restrict__ A,
                                 const float* __restrict__ Bm,
                                 float* __restrict__ C,
                                 const float* __restrict__ bias,
                                 int K, int lda, int ldb, int ldc)
{
    constexpr int BM = 32 * WM_TILES;
    constexpr int BN = 32 * WN_TILES;
    constexpr int NT = WM_TILES * WN_TILES * 32;
    constexpr int KP = 32 + 2;                 // LDS row pad (17 banks, odd)

    __shared__ __bf16 As[BM][KP];              // [m][k]
    __shared__ __bf16 Bs[BN][KP];              // transposed: [n][k]

    const int tid  = threadIdx.x;
    const int wave = tid >> 5;
    const int lane = tid & 31;
    const int lr   = lane & 15;                // row (A) / col (B,C) in tile
    const int hi   = lane >> 4;                // lane-half selector
    const int wm   = (wave / WN_TILES) * 32;
    const int wn   = (wave % WN_TILES) * 32;
    const int m0   = blockIdx.x * BM;
    const int n0   = blockIdx.y * BN;

    v8f acc[2][2] = {};

    for (int k0 = 0; k0 < K; k0 += 32) {
        // stage A tile (BM x 32), coalesced over k
        for (int idx = tid; idx < BM * 32; idx += NT) {
            int r = idx >> 5, c = idx & 31;
            As[r][c] = (__bf16)A[(size_t)(m0 + r) * lda + (k0 + c)];
        }
        // stage B tile (32 x BN) transposed, coalesced over n
        for (int idx = tid; idx < 32 * BN; idx += NT) {
            int kk = idx / BN, nn = idx % BN;
            Bs[nn][kk] = (__bf16)Bm[(size_t)(k0 + kk) * ldb + (n0 + nn)];
        }
        __syncthreads();

        v16bf afrag[2], bfrag[2];
        #pragma unroll
        for (int i = 0; i < 2; ++i) {          // A fragment, ISA 16-bit layout
            const __bf16* row = &As[wm + i * 16 + lr][0];
            const int kb = hi * 8;
            #pragma unroll
            for (int j = 0; j < 8; ++j) afrag[i][j]     = row[kb + j];
            #pragma unroll
            for (int j = 0; j < 8; ++j) afrag[i][8 + j] = row[kb + 16 + j];
        }
        #pragma unroll
        for (int i = 0; i < 2; ++i) {          // B fragment (column of tile)
            const __bf16* colp = &Bs[wn + i * 16 + lr][0];
            const int kb = hi * 8;
            #pragma unroll
            for (int j = 0; j < 8; ++j) bfrag[i][j]     = colp[kb + j];
            #pragma unroll
            for (int j = 0; j < 8; ++j) bfrag[i][8 + j] = colp[kb + 16 + j];
        }
        #pragma unroll
        for (int i = 0; i < 2; ++i)
            #pragma unroll
            for (int j = 0; j < 2; ++j)
                acc[i][j] = __builtin_amdgcn_wmma_f32_16x16x32_bf16(
                    false, afrag[i], false, bfrag[j],
                    (short)0, acc[i][j], false, false);
        __syncthreads();
    }

    // store: D VGPR r -> row = hi*8 + r, col = lr  (ISA C/D layout)
    #pragma unroll
    for (int i = 0; i < 2; ++i) {
        const int mrow = m0 + wm + i * 16 + hi * 8;
        #pragma unroll
        for (int j = 0; j < 2; ++j) {
            const int ncol = n0 + wn + j * 16 + lr;
            #pragma unroll
            for (int r = 0; r < 8; ++r) {
                float v = acc[i][j][r];
                if constexpr (EPI == 1) {      // softplus(v + b_dt[col])
                    v += bias[ncol];
                    v = (v > 20.0f) ? v : log1pf(__expf(v));
                }
                C[(size_t)(mrow + r) * ldc + ncol] = v;
            }
        }
    }
}

// ---------------------------------------------------------------------------
// Depthwise causal conv (DC=4) + bias + SiLU.  xin = xz[..., :DI].
// ---------------------------------------------------------------------------
__global__ void conv_silu_kernel(const float* __restrict__ xz,
                                 const float* __restrict__ w,
                                 const float* __restrict__ bias,
                                 float* __restrict__ xc)
{
    const int idx = blockIdx.x * blockDim.x + threadIdx.x;  // ROWS_*DI_
    const int d = idx & (DI_ - 1);
    const int l = (idx >> 10) & (L_ - 1);
    const int b = idx >> 20;

    const float w0 = w[d * 4 + 0], w1 = w[d * 4 + 1];
    const float w2 = w[d * 4 + 2], w3 = w[d * 4 + 3];
    const size_t base = ((size_t)b * L_ + l) * (2 * DI_) + d;

    float acc = bias[d];
    if (l >= 3) acc += xz[base - 3 * (size_t)(2 * DI_)] * w0;
    if (l >= 2) acc += xz[base - 2 * (size_t)(2 * DI_)] * w1;
    if (l >= 1) acc += xz[base - 1 * (size_t)(2 * DI_)] * w2;
    acc += xz[base] * w3;

    const float s = acc / (1.0f + __expf(-acc));            // SiLU
    xc[((size_t)b * L_ + l) * DI_ + d] = s;
}

// ---------------------------------------------------------------------------
// Selective scan.  One thread per (b,d) channel; 16-state recurrence in VGPRs,
// B_t/C_t broadcast via LDS (x_dbl cols 32..63 are [B|C]).  Fuses the Dp*xc
// skip connection and SiLU(z) gating into the write-out.
// ---------------------------------------------------------------------------
__global__ void scan_kernel(const float* __restrict__ xdbl,
                            const float* __restrict__ dt,
                            const float* __restrict__ xc,
                            const float* __restrict__ xz,
                            const float* __restrict__ A_log,
                            const float* __restrict__ Dp,
                            float* __restrict__ y)
{
    __shared__ float BC[32];
    const int b = blockIdx.y;
    const int d = blockIdx.x * 256 + threadIdx.x;

    float Ac[ST_];
    #pragma unroll
    for (int s = 0; s < ST_; ++s) Ac[s] = -__expf(A_log[d * ST_ + s]);
    const float Dd = Dp[d];

    float h[ST_];
    #pragma unroll
    for (int s = 0; s < ST_; ++s) h[s] = 0.0f;

    for (int t = 0; t < L_; ++t) {
        const size_t row = (size_t)b * L_ + t;
        if (threadIdx.x < 32)
            BC[threadIdx.x] = xdbl[row * (RK_ + 2 * ST_) + RK_ + threadIdx.x];
        __syncthreads();

        const float dtv = dt[row * DI_ + d];
        const float u   = xc[row * DI_ + d];
        const float du  = dtv * u;
        float yv = 0.0f;
        #pragma unroll
        for (int s = 0; s < ST_; ++s) {
            const float dA = __expf(dtv * Ac[s]);
            h[s] = h[s] * dA + du * BC[s];
            yv  += h[s] * BC[ST_ + s];
        }
        const float zv = xz[row * (2 * DI_) + DI_ + d];
        const float g  = zv / (1.0f + __expf(-zv));
        y[row * DI_ + d] = (yv + Dd * u) * g;
        __syncthreads();
    }
}

// ---------------------------------------------------------------------------
// d_out = broadcast(b_fc); must rerun every launch (graph replay, atomics).
// ---------------------------------------------------------------------------
__global__ void init_out_kernel(float* __restrict__ out,
                                const float* __restrict__ b_fc)
{
    const int i = blockIdx.x * 256 + threadIdx.x;
    if (i < B_ * OD_) out[i] = b_fc[i & (OD_ - 1)];
}

// ---------------------------------------------------------------------------
// Split-K WMMA for (16 x 524288) @ (524288 x 128).  8 waves per block, one
// 16x16 N-tile each; K-chunk per block; fp32 atomic accumulation into d_out.
// ---------------------------------------------------------------------------
__global__ void gemm_fc_splitk(const float* __restrict__ A,   // (16, K)
                               const float* __restrict__ Bm,  // (K, 128)
                               float* __restrict__ Cout,      // (16, 128)
                               int K, int KC)
{
    __shared__ __bf16 As[16][34];
    __shared__ __bf16 Bs[128][34];
    const int tid  = threadIdx.x;       // 256
    const int wave = tid >> 5;
    const int lane = tid & 31;
    const int lr   = lane & 15;
    const int hi   = lane >> 4;

    const int kstart = blockIdx.x * KC;
    const int kstop  = (kstart + KC < K) ? (kstart + KC) : K;

    v8f acc = {};
    for (int k0 = kstart; k0 < kstop; k0 += 32) {
        for (int idx = tid; idx < 16 * 32; idx += 256) {
            int r = idx >> 5, c = idx & 31;
            As[r][c] = (__bf16)A[(size_t)r * K + (k0 + c)];
        }
        for (int idx = tid; idx < 32 * 128; idx += 256) {
            int kk = idx >> 7, nn = idx & 127;
            Bs[nn][kk] = (__bf16)Bm[(size_t)(k0 + kk) * OD_ + nn];
        }
        __syncthreads();

        v16bf a, bfr;
        {
            const __bf16* row = &As[lr][0];
            const int kb = hi * 8;
            #pragma unroll
            for (int j = 0; j < 8; ++j) a[j]     = row[kb + j];
            #pragma unroll
            for (int j = 0; j < 8; ++j) a[8 + j] = row[kb + 16 + j];
        }
        {
            const __bf16* colp = &Bs[wave * 16 + lr][0];
            const int kb = hi * 8;
            #pragma unroll
            for (int j = 0; j < 8; ++j) bfr[j]     = colp[kb + j];
            #pragma unroll
            for (int j = 0; j < 8; ++j) bfr[8 + j] = colp[kb + 16 + j];
        }
        acc = __builtin_amdgcn_wmma_f32_16x16x32_bf16(
            false, a, false, bfr, (short)0, acc, false, false);
        __syncthreads();
    }

    #pragma unroll
    for (int r = 0; r < 8; ++r)
        atomicAdd(&Cout[(size_t)(hi * 8 + r) * OD_ + wave * 16 + lr], acc[r]);
}

// ---------------------------------------------------------------------------
extern "C" void kernel_launch(void* const* d_in, const int* in_sizes, int n_in,
                              void* d_out, int out_size, void* d_ws, size_t ws_size,
                              hipStream_t stream)
{
    const float* x      = (const float*)d_in[0];
    const float* W_in   = (const float*)d_in[1];
    const float* conv_w = (const float*)d_in[2];
    const float* conv_b = (const float*)d_in[3];
    const float* W_xprj = (const float*)d_in[4];
    const float* W_dt   = (const float*)d_in[5];
    const float* b_dt   = (const float*)d_in[6];
    const float* A_log  = (const float*)d_in[7];
    const float* Dp     = (const float*)d_in[8];
    const float* W_out  = (const float*)d_in[9];
    const float* W_fc   = (const float*)d_in[10];
    const float* b_fc   = (const float*)d_in[11];
    float* out = (float*)d_out;

    // workspace layout (fp32)
    float* xz   = (float*)d_ws;                       // ROWS_ x 2048
    float* xc   = xz   + (size_t)ROWS_ * 2 * DI_;     // ROWS_ x 1024
    float* xdbl = xc   + (size_t)ROWS_ * DI_;         // ROWS_ x 64
    float* dt   = xdbl + (size_t)ROWS_ * (RK_ + 2 * ST_); // ROWS_ x 1024
    float* ybuf = dt   + (size_t)ROWS_ * DI_;         // ROWS_ x 1024
    float* obuf = ybuf + (size_t)ROWS_ * DI_;         // ROWS_ x 512

    // 1) xz = x @ W_in        (16384 x 2048, K=512)
    gemm_bf16_kernel<2, 4, 0><<<dim3(ROWS_ / 64, (2 * DI_) / 128), 256, 0, stream>>>(
        x, W_in, xz, nullptr, DM_, DM_, 2 * DI_, 2 * DI_);

    // 2) xc = silu(causal_conv(xin) + b)
    conv_silu_kernel<<<(ROWS_ * DI_) / 256, 256, 0, stream>>>(xz, conv_w, conv_b, xc);

    // 3) x_dbl = xc @ W_xproj (16384 x 64, K=1024)
    gemm_bf16_kernel<2, 2, 0><<<dim3(ROWS_ / 64, 1), 128, 0, stream>>>(
        xc, W_xprj, xdbl, nullptr, DI_, DI_, RK_ + 2 * ST_, RK_ + 2 * ST_);

    // 4) dt = softplus(x_dbl[:, :32] @ W_dt + b_dt)  (16384 x 1024, K=32)
    gemm_bf16_kernel<2, 4, 1><<<dim3(ROWS_ / 64, DI_ / 128), 256, 0, stream>>>(
        xdbl, W_dt, dt, b_dt, RK_, RK_ + 2 * ST_, DI_, DI_);

    // 5) selective scan + skip + SiLU(z) gating -> ybuf
    scan_kernel<<<dim3(DI_ / 256, B_), 256, 0, stream>>>(
        xdbl, dt, xc, xz, A_log, Dp, ybuf);

    // 6) obuf = y @ W_out     (16384 x 512, K=1024)
    gemm_bf16_kernel<2, 4, 0><<<dim3(ROWS_ / 64, DM_ / 128), 256, 0, stream>>>(
        ybuf, W_out, obuf, nullptr, DI_, DI_, DM_, DM_);

    // 7) d_out = b_fc (re-init every launch; atomics accumulate on top)
    init_out_kernel<<<(B_ * OD_ + 255) / 256, 256, 0, stream>>>(out, b_fc);

    // 8) d_out += reshape(obuf) @ W_fc   (split-K, K=524288, 128 blocks)
    gemm_fc_splitk<<<128, 256, 0, stream>>>(obuf, W_fc, out, L_ * DM_, 4096);
}
